// MultiHeadSelfAttention_86431921865284
// MI455X (gfx1250) — matrix operand order repaired and checked by reading, not compiled, not run on hardware
//
#include <hip/hip_runtime.h>
#include <hip/hip_bf16.h>

// ---------------- problem constants ----------------
constexpr int B_  = 2;
constexpr int S_  = 2048;
constexpr int D_  = 1024;
constexpr int H_  = 16;
constexpr int HD_ = 64;            // head dim
constexpr int NT_ = B_ * S_;       // 4096 tokens

// ---------------- WMMA types ----------------
typedef __attribute__((ext_vector_type(16))) __bf16 v16bf;
typedef __attribute__((ext_vector_type(8)))  __bf16 v8bf;
typedef __attribute__((ext_vector_type(8)))  float  v8f;
typedef __attribute__((ext_vector_type(4)))  int    v4i;

__device__ inline v16bf cat8(v8bf lo, v8bf hi) {
  return __builtin_shufflevector(lo, hi, 0,1,2,3,4,5,6,7,8,9,10,11,12,13,14,15);
}

// 16x32 bf16 A fragment, source row-major with row stride ld (elements).
// Contiguous-K layout used here: lane m (=lane&15) holds row M=m; hi half selects K offset:
//   elements 0..15 -> K = 16*hi + e
__device__ inline v16bf load_afrag(const __bf16* tile, int ld) {
  const int lane = threadIdx.x & 31;
  const int m = lane & 15, hi = lane >> 4;
  const __bf16* r = tile + (size_t)m * ld + 16 * hi;
  v8bf lo = *(const v8bf*)(r);
  v8bf hh = *(const v8bf*)(r + 8);
  return cat8(lo, hh);
}

// 32x16 bf16 B fragment. Source stored K-contiguous per column (column n has
// stride ldn elements). Contiguous-K layout used here (same as A): lane n holds column N=n; K = 16*hi + e.
__device__ inline v16bf load_bfrag(const __bf16* tile, int ldn) {
  const int lane = threadIdx.x & 31;
  const int n = lane & 15, hi = lane >> 4;
  return *(const v16bf*)(tile + (size_t)n * ldn + 16 * hi);
}

__device__ inline v8f wmma_bf16(v16bf a, v16bf b, v8f c) {
  return __builtin_amdgcn_wmma_f32_16x16x32_bf16(false, a, false, b,
                                                 (short)0, c, false, false);
}

// ---------------- gfx1250 async global->LDS helpers ----------------
__device__ inline void async_copy16(const __bf16* g, __bf16* l) {
#if __has_builtin(__builtin_amdgcn_global_load_async_to_lds_b128)
  __builtin_amdgcn_global_load_async_to_lds_b128(
      (__attribute__((address_space(1))) v4i*)g,
      (__attribute__((address_space(3))) v4i*)l, 0, 0);
#else
  *(v8bf*)l = *(const v8bf*)g;   // fallback: load + ds_store
#endif
}

__device__ inline void wait_async0() {
#if __has_builtin(__builtin_amdgcn_s_wait_asynccnt)
  __builtin_amdgcn_s_wait_asynccnt(0);
#else
  asm volatile("s_wait_asynccnt 0x0" ::: "memory");
#endif
}

__device__ inline void wait_ds0() {
  asm volatile("s_wait_dscnt 0x0" ::: "memory");
}

// ---------------- conversion kernels ----------------
__global__ void cvt_cast_f32_bf16(const float* __restrict__ in,
                                  __bf16* __restrict__ out, int n) {
  int i = blockIdx.x * blockDim.x + threadIdx.x;
  if (i < n) out[i] = (__bf16)in[i];
}

// in: [D][D] (k,n) row-major fp32 -> out: [n][k] bf16 (transposed)
__global__ void cvt_transpose_bf16(const float* __restrict__ in,
                                   __bf16* __restrict__ out) {
  int i = blockIdx.x * blockDim.x + threadIdx.x;   // over D*D
  int nn = i / D_, kk = i % D_;
  out[i] = (__bf16)in[(size_t)kk * D_ + nn];
}

// Stage a 64col x 32k bf16 weight tile into LDS (async). 256 threads,
// each moves one 16B chunk; tile stored [col][k] with ldn=32.
__device__ inline void stage_w_tile(const __bf16* __restrict__ wt, int c0,
                                    int k0, __bf16* buf) {
  const int tid = threadIdx.x;     // 0..255
  const int col = tid >> 2;        // 0..63
  const int sub = tid & 3;         // 0..3  (16B chunk within 32 elements)
  const __bf16* g = wt + (size_t)(c0 + col) * D_ + k0 + sub * 8;
  async_copy16(g, buf + col * 32 + sub * 8);
}

// ---------------- cooperative GEMM core ----------------
// Block: 256 threads = 8 waves. Block tile: 128 tokens x 64 cols.
// Weight tile double-buffered in LDS via async copies; each wave computes
// a 16x64 slice with 4 f32 accumulators.
template <typename StoreFn>
__device__ inline void gemm_core(const __bf16* __restrict__ x,
                                 const __bf16* __restrict__ wt,
                                 int c0, int tblk, __bf16 (*wbuf)[64 * 32],
                                 StoreFn&& store) {
  const int wave = threadIdx.x >> 5;
  const int t0 = tblk + wave * 16;

  // prologue: stage k0=0
  stage_w_tile(wt, c0, 0, wbuf[0]);
  wait_async0();
  wait_ds0();
  __syncthreads();

  v8f acc[4] = {};
  v16bf a = load_afrag(x + (size_t)t0 * D_, D_);
  int cur = 0;
  for (int k0 = 0; k0 < D_; k0 += 32) {
    const int nxt = cur ^ 1;
    if (k0 + 32 < D_) stage_w_tile(wt, c0, k0 + 32, wbuf[nxt]);

    // prefetch next A fragment while computing this step
    v16bf anext = a;
    if (k0 + 32 < D_) anext = load_afrag(x + (size_t)t0 * D_ + k0 + 32, D_);

    v16bf bfr[4];
#pragma unroll
    for (int n4 = 0; n4 < 4; ++n4)
      bfr[n4] = load_bfrag(wbuf[cur] + 16 * n4 * 32, 32);
#pragma unroll
    for (int n4 = 0; n4 < 4; ++n4) acc[n4] = wmma_bf16(a, bfr[n4], acc[n4]);

    a = anext;
    wait_async0();   // staged tile for next step complete (this wave's share)
    wait_ds0();      // this wave's LDS reads of `cur` landed in VGPRs
    __syncthreads(); // all waves done: safe to consume nxt / overwrite cur
    cur = nxt;
  }
  store(acc, t0);
}

// ---------------- fused QKV projection ----------------
// Q,K stored [B,H,S,HD]; V stored transposed [B,H,HD,S]. z selects Q/K/V.
__global__ __launch_bounds__(256)
void qkv_gemm(const __bf16* __restrict__ x,
              const __bf16* __restrict__ wqt, const __bf16* __restrict__ wkt,
              const __bf16* __restrict__ wvt,
              const float* __restrict__ bq, const float* __restrict__ bk,
              const float* __restrict__ bv,
              __bf16* __restrict__ qo, __bf16* __restrict__ ko,
              __bf16* __restrict__ vto) {
  __shared__ __bf16 wbuf[2][64 * 32];
  const int c0 = blockIdx.x * 64;
  const int tblk = blockIdx.y * 128;
  const int z = blockIdx.z;
  const __bf16* wt   = (z == 0) ? wqt : (z == 1) ? wkt : wvt;
  const float*  bias = (z == 0) ? bq  : (z == 1) ? bk  : bv;

  gemm_core(x, wt, c0, tblk, wbuf, [&](v8f (&acc)[4], int t0) {
    const int lane = threadIdx.x & 31;
    const int n = lane & 15, hi = lane >> 4;
    const int b = t0 / S_, s0 = t0 % S_;
#pragma unroll
    for (int n4 = 0; n4 < 4; ++n4) {
      const int col = c0 + 16 * n4 + n;
      const int h = col >> 6, hd = col & 63;
      const float bb = bias[col];
      const size_t bh = (size_t)b * H_ + h;
#pragma unroll
      for (int r = 0; r < 8; ++r) {
        const int s = s0 + r + 8 * hi;
        const __bf16 o = (__bf16)(acc[n4][r] + bb);
        if (z == 2)      vto[(bh * HD_ + hd) * S_ + s] = o;
        else if (z == 1) ko[(bh * S_ + s) * HD_ + hd] = o;
        else             qo[(bh * S_ + s) * HD_ + hd] = o;
      }
    }
  });
}

// ---------------- output projection ----------------
__global__ __launch_bounds__(256)
void out_proj(const __bf16* __restrict__ o, const __bf16* __restrict__ wot,
              const float* __restrict__ bo, float* __restrict__ out) {
  __shared__ __bf16 wbuf[2][64 * 32];
  const int c0 = blockIdx.x * 64;
  const int tblk = blockIdx.y * 128;

  gemm_core(o, wot, c0, tblk, wbuf, [&](v8f (&acc)[4], int t0) {
    const int lane = threadIdx.x & 31;
    const int n = lane & 15, hi = lane >> 4;
#pragma unroll
    for (int n4 = 0; n4 < 4; ++n4) {
      const int col = c0 + 16 * n4 + n;
      const float bb = bo[col];
#pragma unroll
      for (int r = 0; r < 8; ++r)
        out[(size_t)(t0 + r + 8 * hi) * D_ + col] = acc[n4][r] + bb;
    }
  });
}

// ---------------- flash attention ----------------
// One wave per (b*h, 16-query tile). Online softmax over 32-key steps.
__global__ __launch_bounds__(32)
void attn_fwd(const __bf16* __restrict__ q, const __bf16* __restrict__ k,
              const __bf16* __restrict__ vt, const int* __restrict__ mask,
              __bf16* __restrict__ o) {
  __shared__ __bf16 pLds[16 * 32];
  const int lane = threadIdx.x;
  const int q0 = blockIdx.x * 16;
  const int bh = blockIdx.y;
  const int b  = bh / H_;
  const float scale = 0.125f;  // 1/sqrt(64)

  const __bf16* qptr = q  + (size_t)bh * S_ * HD_;
  const __bf16* kptr = k  + (size_t)bh * S_ * HD_;
  const __bf16* vptr = vt + (size_t)bh * HD_ * S_;

  const v16bf qa0 = load_afrag(qptr + (size_t)q0 * HD_ + 0,  HD_);
  const v16bf qa1 = load_afrag(qptr + (size_t)q0 * HD_ + 32, HD_);

  const int n = lane & 15, hi = lane >> 4;
  float mrow[8], lrow[8];
  v8f acc[4] = {};
#pragma unroll
  for (int r = 0; r < 8; ++r) { mrow[r] = -1e30f; lrow[r] = 0.f; }

  const int nk = q0 + 16;  // causal horizon for this tile
  for (int t0 = 0; t0 < nk; t0 += 32) {
    // ---- hoist all global loads for this step: 4 K frags + 4 V frags
    v16bf kf[4], vb[4];
#pragma unroll
    for (int j = 0; j < 2; ++j) {
      kf[2 * j + 0] = load_bfrag(kptr + (size_t)(t0 + 16 * j) * HD_ + 0,  HD_);
      kf[2 * j + 1] = load_bfrag(kptr + (size_t)(t0 + 16 * j) * HD_ + 32, HD_);
    }
#pragma unroll
    for (int n4 = 0; n4 < 4; ++n4)
      vb[n4] = load_bfrag(vptr + (size_t)(16 * n4) * S_ + t0, S_);

    // ---- scores: two 16x16 tiles over 32 keys, K=HD=64 -> 2 WMMAs each
    v8f sc[2];
#pragma unroll
    for (int j = 0; j < 2; ++j) {
      v8f c = {};
      c = wmma_bf16(qa0, kf[2 * j + 0], c);
      c = wmma_bf16(qa1, kf[2 * j + 1], c);
      sc[j] = c;
    }

    int  key[2]; bool kok[2];
#pragma unroll
    for (int j = 0; j < 2; ++j) {
      key[j] = t0 + 16 * j + n;
      kok[j] = (mask[(size_t)b * S_ + key[j]] == 0);
    }

    float tmax[8];
#pragma unroll
    for (int r = 0; r < 8; ++r) {
      const int qrow = q0 + r + 8 * hi;
      const float v0 = (kok[0] && key[0] <= qrow) ? sc[0][r] * scale : -1e30f;
      const float v1 = (kok[1] && key[1] <= qrow) ? sc[1][r] * scale : -1e30f;
      sc[0][r] = v0; sc[1][r] = v1;
      tmax[r] = fmaxf(v0, v1);
    }
#pragma unroll
    for (int m2 = 1; m2 < 16; m2 <<= 1)
#pragma unroll
      for (int r = 0; r < 8; ++r)
        tmax[r] = fmaxf(tmax[r], __shfl_xor(tmax[r], m2, 32));

    float corr[8], rsum[8];
#pragma unroll
    for (int r = 0; r < 8; ++r) {
      const float mn = fmaxf(mrow[r], tmax[r]);
      corr[r] = __expf(mrow[r] - mn);
      mrow[r] = mn;
      const int qrow = q0 + r + 8 * hi;
      const float p0 = (kok[0] && key[0] <= qrow) ? __expf(sc[0][r] - mn) : 0.f;
      const float p1 = (kok[1] && key[1] <= qrow) ? __expf(sc[1][r] - mn) : 0.f;
      rsum[r] = p0 + p1;
      // stage P (C/D layout) into LDS for re-read in A layout
      pLds[(r + 8 * hi) * 32 + n]      = (__bf16)p0;
      pLds[(r + 8 * hi) * 32 + 16 + n] = (__bf16)p1;
    }
#pragma unroll
    for (int m2 = 1; m2 < 16; m2 <<= 1)
#pragma unroll
      for (int r = 0; r < 8; ++r)
        rsum[r] += __shfl_xor(rsum[r], m2, 32);
#pragma unroll
    for (int r = 0; r < 8; ++r) lrow[r] = lrow[r] * corr[r] + rsum[r];
#pragma unroll
    for (int n4 = 0; n4 < 4; ++n4)
#pragma unroll
      for (int r = 0; r < 8; ++r) acc[n4][r] *= corr[r];

    // wave-local LDS RAW fence (CDNA5 split counter), then P as A-fragment
    wait_ds0();
    const v16bf pa = load_afrag(&pLds[0], 32);

    // acc += P(16x32) x V(32x64); V transposed so lane loads are contiguous
#pragma unroll
    for (int n4 = 0; n4 < 4; ++n4) acc[n4] = wmma_bf16(pa, vb[n4], acc[n4]);
  }

  // ---- normalize + write [B,S,H*HD] bf16
  const int h = bh % H_;
#pragma unroll
  for (int n4 = 0; n4 < 4; ++n4) {
    const int col = h * HD_ + 16 * n4 + n;
#pragma unroll
    for (int r = 0; r < 8; ++r) {
      const float denom = (lrow[r] == 0.f) ? 1.f : lrow[r];
      const size_t token = (size_t)b * S_ + q0 + r + 8 * hi;
      o[token * D_ + col] = (__bf16)(acc[n4][r] / denom);
    }
  }
}

// ---------------- launch ----------------
extern "C" void kernel_launch(void* const* d_in, const int* in_sizes, int n_in,
                              void* d_out, int out_size, void* d_ws, size_t ws_size,
                              hipStream_t stream) {
  const float* inputs = (const float*)d_in[0];
  const int*   amask  = (const int*)d_in[1];
  const float* Wq = (const float*)d_in[2];
  const float* bq = (const float*)d_in[3];
  const float* Wk = (const float*)d_in[4];
  const float* bk = (const float*)d_in[5];
  const float* Wv = (const float*)d_in[6];
  const float* bv = (const float*)d_in[7];
  const float* Wo = (const float*)d_in[8];
  const float* bo = (const float*)d_in[9];
  float* out = (float*)d_out;

  // workspace carve-up (bf16 elements)
  __bf16* base = (__bf16*)d_ws;
  __bf16* xbf = base;                               // NT*D
  __bf16* wqt = xbf + (size_t)NT_ * D_;             // D*D
  __bf16* wkt = wqt + (size_t)D_ * D_;
  __bf16* wvt = wkt + (size_t)D_ * D_;
  __bf16* wot = wvt + (size_t)D_ * D_;
  __bf16* qm  = wot + (size_t)D_ * D_;              // NT*D as [B,H,S,HD]
  __bf16* km  = qm  + (size_t)NT_ * D_;
  __bf16* vtm = km  + (size_t)NT_ * D_;             // [B,H,HD,S]
  __bf16* obf = vtm + (size_t)NT_ * D_;             // NT*D

  // 1) casts
  {
    const int n = NT_ * D_;
    cvt_cast_f32_bf16<<<(n + 255) / 256, 256, 0, stream>>>(inputs, xbf, n);
    const int nw = D_ * D_;
    cvt_transpose_bf16<<<(nw + 255) / 256, 256, 0, stream>>>(Wq, wqt);
    cvt_transpose_bf16<<<(nw + 255) / 256, 256, 0, stream>>>(Wk, wkt);
    cvt_transpose_bf16<<<(nw + 255) / 256, 256, 0, stream>>>(Wv, wvt);
    cvt_transpose_bf16<<<(nw + 255) / 256, 256, 0, stream>>>(Wo, wot);
  }

  // 2) fused QKV projection (cooperative, async-LDS double-buffered)
  qkv_gemm<<<dim3(D_ / 64, NT_ / 128, 3), 256, 0, stream>>>(
      xbf, wqt, wkt, wvt, bq, bk, bv, qm, km, vtm);

  // 3) flash attention
  attn_fwd<<<dim3(S_ / 16, B_ * H_), 32, 0, stream>>>(qm, km, vtm, amask, obf);

  // 4) output projection
  out_proj<<<dim3(D_ / 64, NT_ / 128), 256, 0, stream>>>(obf, wot, bo, out);
}
